// Attention_67662914781591
// MI455X (gfx1250) — matrix operand order repaired and checked by reading, not compiled, not run on hardware
//
#include <hip/hip_runtime.h>

typedef unsigned short u16;
typedef __attribute__((ext_vector_type(16))) __bf16 v16bf;
typedef __attribute__((ext_vector_type(8)))  float  v8f;
typedef __attribute__((ext_vector_type(4)))  unsigned tdm_g0_t;
typedef __attribute__((ext_vector_type(8)))  int      tdm_g1_t;
typedef __attribute__((ext_vector_type(4)))  int      tdm_g23_t;
typedef __attribute__((ext_vector_type(8)))  int      tdm_x_t;

union FragAB { v16bf v; uint4 q[2]; };
union FragC  { v8f   v; float f[8]; };

__device__ __forceinline__ u16 f2bf(float f) {
    unsigned u = __builtin_bit_cast(unsigned, f);
    u += 0x7FFFu + ((u >> 16) & 1u);            // round-to-nearest-even
    return (u16)(u >> 16);
}
__device__ __forceinline__ float bf2f(u16 u) {
    unsigned x = ((unsigned)u) << 16;
    return __builtin_bit_cast(float, x);
}
// LDS byte offset of a generic pointer into shared memory
__device__ __forceinline__ unsigned lds_addr_of(void* p) {
    return (unsigned)(size_t)(__attribute__((address_space(3))) void*)p;
}

// ---------------------------------------------------------------------------
// TDM: async DMA of a 2-D fp32 tile (rows x cols, row stride in elements)
// from global memory into LDS, with LDS row padding so each 128-dword row
// lands at a 132-dword (bank-conflict-free) stride.  D# per CDNA5 ISA §8.
// ---------------------------------------------------------------------------
__device__ __forceinline__ void tdm_load_2d_f32(
    const void* gaddr, unsigned lds_byte_off,
    unsigned cols, unsigned rows, unsigned row_stride_elems)
{
    unsigned long long ga = (unsigned long long)(size_t)gaddr;
    tdm_g0_t g0;
    g0[0] = 1u;                                   // count=1 (valid user D#)
    g0[1] = lds_byte_off;                         // lds_addr [63:32]
    g0[2] = (unsigned)(ga & 0xFFFFFFFFu);         // global_addr [95:64]
    g0[3] = (unsigned)((ga >> 32) & 0x01FFFFFFu)  // global_addr [120:96]
          | (2u << 30);                           // type=2 (image)
    tdm_g1_t g1;
    // data_size=4B, pad_enable, pad_interval=6 (128 dwords), pad_amount=3 (+4 dwords)
    g1[0] = (int)((2u << 16) | (1u << 20) | (6u << 22) | (3u << 25));
    g1[1] = (int)((cols & 0xFFFFu) << 16);        // tensor_dim0[15:0] (==tile)
    g1[2] = (int)(((cols >> 16) & 0xFFFFu) | ((rows & 0xFFFFu) << 16)); // dim0 hi | dim1 lo
    g1[3] = (int)(((rows >> 16) & 0xFFFFu) | ((cols & 0xFFFFu) << 16)); // dim1 hi | tile_dim0
    g1[4] = (int)(rows & 0xFFFFu);                // tile_dim1 | tile_dim2=0
    g1[5] = (int)row_stride_elems;                // tensor_dim0_stride[31:0]
    g1[6] = 0;
    g1[7] = 0;
    tdm_g23_t z4 = {0, 0, 0, 0};
    tdm_x_t   z8 = {0, 0, 0, 0, 0, 0, 0, 0};
    __builtin_amdgcn_tensor_load_to_lds(g0, g1, z4, z4, z8, 0);
}

// ---------------------------------------------------------------------------
// Generic C(fp32) = A(M x K) @ B(K x N), bf16 WMMA.  A row-major (fp32 or
// bf16 source), B row-major bf16.  Workgroup tile 128x128, 8 waves, each wave
// owns a 32x64 sub-tile (2 M-tiles x 4 N-tiles).  k-step = 32.
// LDS strides padded to 40 bf16 (80 B): 16-lane b128 reads are bank-clean.
// ---------------------------------------------------------------------------
template<bool A_F32>
__global__ __launch_bounds__(256) void gemm_bf16(
    const void* __restrict__ Ain, const u16* __restrict__ B,
    float* __restrict__ C, int lda, int ldb, int ldc, int K)
{
    __shared__ u16 sA[128 * 40];
    __shared__ u16 sBT[128 * 40];   // transposed: [n][k]

    const int tid  = threadIdx.x;
    const int wave = tid >> 5, lane = tid & 31;
    const int half = lane >> 4, l16 = lane & 15;
    const int m0 = blockIdx.x * 128, n0 = blockIdx.y * 128;
    const int wm = (wave & 3) * 32;
    const int wn = (wave >> 2) * 64;

    FragC acc[2][4];
#pragma unroll
    for (int i = 0; i < 2; i++)
#pragma unroll
        for (int j = 0; j < 4; j++)
#pragma unroll
            for (int r = 0; r < 8; r++) acc[i][j].f[r] = 0.f;

    for (int k0 = 0; k0 < K; k0 += 32) {
        // L2 prefetch of next k-tile (global_prefetch_b8)
        if (k0 + 32 < K) {
            __builtin_prefetch(B + (size_t)(k0 + 32 + (tid >> 3)) * ldb + n0 + ((tid & 7) << 4), 0, 1);
            if (A_F32)
                __builtin_prefetch((const float*)Ain + (size_t)(m0 + (tid >> 1)) * lda + k0 + 32 + (tid & 1) * 16, 0, 1);
        }
        // ---- stage A tile: 128 rows x 32 k, bf16 ----
        if (A_F32) {
            const float* A = (const float*)Ain;
#pragma unroll
            for (int i = 0; i < 4; i++) {
                int idx = tid + i * 256;               // 0..1023
                int r = idx >> 3, c4 = (idx & 7) << 2;
                float4 v = *(const float4*)(A + (size_t)(m0 + r) * lda + k0 + c4);
                u16* d = &sA[r * 40 + c4];
                d[0] = f2bf(v.x); d[1] = f2bf(v.y); d[2] = f2bf(v.z); d[3] = f2bf(v.w);
            }
        } else {
            const u16* A = (const u16*)Ain;
#pragma unroll
            for (int i = 0; i < 4; i++) {
                int idx = tid + i * 256;
                int r = idx >> 3, c4 = (idx & 7) << 2;
                *(uint2*)&sA[r * 40 + c4] =
                    *(const uint2*)(A + (size_t)(m0 + r) * lda + k0 + c4);
            }
        }
        // ---- stage B tile transposed: 32 k x 128 n -> sBT[n][k] ----
#pragma unroll
        for (int i = 0; i < 4; i++) {
            int idx = tid + i * 256;                   // 0..1023
            int kr = idx >> 5, c4 = (idx & 31) << 2;
            uint2 v = *(const uint2*)(B + (size_t)(k0 + kr) * ldb + n0 + c4);
            sBT[(c4 + 0) * 40 + kr] = (u16)(v.x);
            sBT[(c4 + 1) * 40 + kr] = (u16)(v.x >> 16);
            sBT[(c4 + 2) * 40 + kr] = (u16)(v.y);
            sBT[(c4 + 3) * 40 + kr] = (u16)(v.y >> 16);
        }
        __syncthreads();

        FragAB a[2], b[4];
#pragma unroll
        for (int i = 0; i < 2; i++) {
            const u16* p = &sA[(wm + i * 16 + l16) * 40 + half * 8];
            a[i].q[0] = *(const uint4*)p;
            a[i].q[1] = *(const uint4*)(p + 16);
        }
#pragma unroll
        for (int j = 0; j < 4; j++) {
            const u16* p = &sBT[(wn + j * 16 + l16) * 40 + half * 8];
            b[j].q[0] = *(const uint4*)p;
            b[j].q[1] = *(const uint4*)(p + 16);
        }
#pragma unroll
        for (int i = 0; i < 2; i++)
#pragma unroll
            for (int j = 0; j < 4; j++)
                acc[i][j].v = __builtin_amdgcn_wmma_f32_16x16x32_bf16(
                    false, a[i].v, false, b[j].v, (short)0, acc[i][j].v, false, false);
        __syncthreads();
    }

    // ---- epilogue: fp32 store ----
#pragma unroll
    for (int i = 0; i < 2; i++) {
        int row0 = m0 + wm + i * 16 + half * 8;
#pragma unroll
        for (int j = 0; j < 4; j++) {
            int col = n0 + wn + j * 16 + l16;
#pragma unroll
            for (int r = 0; r < 8; r++)
                C[(size_t)(row0 + r) * ldc + col] = acc[i][j].f[r];
        }
    }
}

// ---------------------------------------------------------------------------
// Rotary (fp32, faithful to reference) + bf16 downconvert of q/k/v.
// qkv: 512 x 6144 fp32 (q | k | v);  freqs: 16 x 64 complex64 (cos,sin pairs).
// ---------------------------------------------------------------------------
__global__ __launch_bounds__(256) void rotary_convert(
    const float* __restrict__ qkv, const float* __restrict__ freqs,
    u16* __restrict__ qb, u16* __restrict__ kb, u16* __restrict__ vb)
{
    int t = blockIdx.x * 256 + threadIdx.x;   // 512*3072 work items (pairs)
    int m = t / 3072, rest = t % 3072;
    int qpos = m & 15;
    const float2* fc = (const float2*)freqs;
    if (rest < 2048) {                        // q: pair p -> cols 2p,2p+1
        int p = rest, i = p & 63;
        float2 f = fc[qpos * 64 + i];
        float a = qkv[(size_t)m * 6144 + 2 * p];
        float b = qkv[(size_t)m * 6144 + 2 * p + 1];
        qb[(size_t)m * 4096 + 2 * p]     = f2bf(a * f.x - b * f.y);
        qb[(size_t)m * 4096 + 2 * p + 1] = f2bf(a * f.y + b * f.x);
    } else if (rest < 2560) {                 // k
        int p = rest - 2048, i = p & 63;
        float2 f = fc[qpos * 64 + i];
        float a = qkv[(size_t)m * 6144 + 4096 + 2 * p];
        float b = qkv[(size_t)m * 6144 + 4096 + 2 * p + 1];
        kb[(size_t)m * 1024 + 2 * p]     = f2bf(a * f.x - b * f.y);
        kb[(size_t)m * 1024 + 2 * p + 1] = f2bf(a * f.y + b * f.x);
    } else {                                  // v: plain convert
        int p = rest - 2560;
        vb[(size_t)m * 1024 + 2 * p]     = f2bf(qkv[(size_t)m * 6144 + 5120 + 2 * p]);
        vb[(size_t)m * 1024 + 2 * p + 1] = f2bf(qkv[(size_t)m * 6144 + 5120 + 2 * p + 1]);
    }
}

// ---------------------------------------------------------------------------
// Flash attention over the 4096-entry cache.  One workgroup per (kv-head, b):
// 4 GQA query heads (64 q-rows) share each streamed KV chunk -> each fp32
// cache byte is read exactly once.  32 chunks x 128 keys.
// KV chunks are DMA'd by the Tensor Data Mover into LDS (software pipelined:
// chunk ch+1 is in flight while chunk ch computes).  Last 16 positions come
// from freshly computed k/v (reference's cache append, inputs not mutated).
// Online softmax; causal mask only touches chunk 31.
//
// LDS map (dynamic, 250.75 KB):
//   [0)       sKf  fp32 [128][132]  TDM dest, K chunk     67584 B
//   [67584)   sVf  fp32 [128][132]  TDM dest, V chunk     67584 B
//   [135168)  sK   bf16 [128 keys][136]                   34816 B
//   [169984)  sVT  bf16 [128 dims][136] (transposed)      34816 B
//   [204800)  sP   bf16 [64 rows][136]                    17408 B
//   [222208)  sS   fp32 [64 rows][132]                    33792 B
//   [256000)  sMax/sSum/sAlpha fp32 [64] each               768 B
// ---------------------------------------------------------------------------
#define ATTN_LDS_BYTES 256768

__global__ __launch_bounds__(256) void attn_fwd(
    const float* __restrict__ cache_k, const float* __restrict__ cache_v,
    const u16* __restrict__ qb, const u16* __restrict__ kb,
    const u16* __restrict__ vb, u16* __restrict__ ob)
{
    extern __shared__ char smem[];
    float* sKf   = (float*)smem;
    float* sVf   = (float*)(smem + 67584);
    u16*   sK    = (u16*)(smem + 135168);
    u16*   sVT   = (u16*)(smem + 169984);
    u16*   sP    = (u16*)(smem + 204800);
    float* sS    = (float*)(smem + 222208);
    float* sMax  = (float*)(smem + 256000);
    float* sSum  = sMax + 64;
    float* sAlpha= sMax + 128;

    const int tid = threadIdx.x, wave = tid >> 5, lane = tid & 31;
    const int half = lane >> 4, l16 = lane & 15;
    const int kv = blockIdx.x;                  // 0..7
    const int b  = blockIdx.y;                  // 0..31
    const int hloc = wave >> 1;                 // 0..3 local q-head
    const int h = kv * 4 + hloc;
    const int wsel = wave & 1;                  // key-tile / dim-tile half
    const float scale = 0.08838834764831845f;   // 1/sqrt(128)

    const unsigned kf_off = lds_addr_of(sKf);
    const unsigned vf_off = lds_addr_of(sVf);
    const size_t cache_base = ((size_t)b * 4096 * 8 + kv) * 128;

    // persistent Q fragments for this wave's head (16 x 128, 4 k-steps)
    FragAB qf[4];
    {
        const u16* qrow = qb + (size_t)(b * 16 + l16) * 4096 + h * 128;
#pragma unroll
        for (int ks = 0; ks < 4; ks++) {
            qf[ks].q[0] = *(const uint4*)(qrow + ks * 32 + half * 8);
            qf[ks].q[1] = *(const uint4*)(qrow + ks * 32 + 16 + half * 8);
        }
    }

    FragC oacc[4];
#pragma unroll
    for (int j = 0; j < 4; j++)
#pragma unroll
        for (int r = 0; r < 8; r++) oacc[j].f[r] = 0.f;
    if (tid < 64) { sMax[tid] = -3.0e38f; sSum[tid] = 0.f; }

    // prologue: TDM chunk 0 in flight
    if (wave == 0) {
        tdm_load_2d_f32(cache_k + cache_base, kf_off, 128, 128, 1024);
        tdm_load_2d_f32(cache_v + cache_base, vf_off, 128, 128, 1024);
    }
    __syncthreads();

    for (int ch = 0; ch < 32; ch++) {
        const int base = ch * 128;
        if (wave == 0) __builtin_amdgcn_s_wait_tensorcnt(0);
        __syncthreads();

        // ---- convert pass: fp32 LDS -> bf16 K (row-major) + V^T; splice the
        //      16 freshly-computed rows over stale cache rows >= 4080 ----
#pragma unroll 4
        for (int i = 0; i < 16; i++) {
            int idx = tid + i * 256;              // 0..4095
            int key = idx >> 5, d4 = (idx & 31) << 2;
            int pos = base + key;
            float4 kvec = *(const float4*)&sKf[key * 132 + d4];
            float4 vvec = *(const float4*)&sVf[key * 132 + d4];
            if (pos >= 4080) {
                size_t m = (size_t)(b * 16 + (pos - 4080)) * 1024 + kv * 128 + d4;
                kvec = make_float4(bf2f(kb[m]), bf2f(kb[m+1]), bf2f(kb[m+2]), bf2f(kb[m+3]));
                vvec = make_float4(bf2f(vb[m]), bf2f(vb[m+1]), bf2f(vb[m+2]), bf2f(vb[m+3]));
            }
            u16* dk = &sK[key * 136 + d4];
            dk[0] = f2bf(kvec.x); dk[1] = f2bf(kvec.y);
            dk[2] = f2bf(kvec.z); dk[3] = f2bf(kvec.w);
            sVT[(d4 + 0) * 136 + key] = f2bf(vvec.x);
            sVT[(d4 + 1) * 136 + key] = f2bf(vvec.y);
            sVT[(d4 + 2) * 136 + key] = f2bf(vvec.z);
            sVT[(d4 + 3) * 136 + key] = f2bf(vvec.w);
        }
        __syncthreads();

        // fp32 staging buffers are free again: kick off next chunk's DMA
        if (wave == 0 && ch + 1 < 32) {
            size_t off = cache_base + (size_t)(base + 128) * 1024;
            tdm_load_2d_f32(cache_k + off, kf_off, 128, 128, 1024);
            tdm_load_2d_f32(cache_v + off, vf_off, 128, 128, 1024);
        }

        // ---- S = Q @ K^T for this wave: head hloc, key-tiles wsel*4..+3 ----
        FragC sacc[4];
#pragma unroll
        for (int j = 0; j < 4; j++)
#pragma unroll
            for (int r = 0; r < 8; r++) sacc[j].f[r] = 0.f;
#pragma unroll
        for (int ks = 0; ks < 4; ks++) {
#pragma unroll
            for (int j = 0; j < 4; j++) {
                int nt = wsel * 4 + j;
                const u16* p = &sK[(nt * 16 + l16) * 136 + ks * 32 + half * 8];
                FragAB kf;
                kf.q[0] = *(const uint4*)p;
                kf.q[1] = *(const uint4*)(p + 16);
                sacc[j].v = __builtin_amdgcn_wmma_f32_16x16x32_bf16(
                    false, qf[ks].v, false, kf.v, (short)0, sacc[j].v, false, false);
            }
        }
#pragma unroll
        for (int j = 0; j < 4; j++) {
            int col = wsel * 64 + j * 16 + l16;
            float* dst = &sS[(hloc * 16 + half * 8) * 132 + col];
#pragma unroll
            for (int r = 0; r < 8; r++) dst[r * 132] = sacc[j].f[r] * scale;
        }
        __syncthreads();

        // ---- online softmax: one thread per q-row (64 rows) ----
        if (tid < 64) {
            int r = tid, qpos = r & 15;
            int lim = 4080 + qpos - base;                 // valid c <= lim
            int cmax = lim < 127 ? lim : 127;             // always >= 112
            const float* sr = &sS[r * 132];
            float mx = sMax[r];
            for (int c = 0; c <= cmax; c++) mx = fmaxf(mx, sr[c]);
            float alpha = __expf(sMax[r] - mx);
            float sum = sSum[r] * alpha;
            u16* pr = &sP[r * 136];
            for (int c = 0; c < 128; c++) {
                float e = (c <= cmax) ? __expf(sr[c] - mx) : 0.f;
                sum += e;
                pr[c] = f2bf(e);
            }
            sMax[r] = mx; sSum[r] = sum; sAlpha[r] = alpha;
        }
        __syncthreads();

        // ---- O = alpha*O + P @ V : head hloc, dim-tiles wsel*4..+3 ----
        float al[8];
#pragma unroll
        for (int r = 0; r < 8; r++) al[r] = sAlpha[hloc * 16 + half * 8 + r];
#pragma unroll
        for (int j = 0; j < 4; j++)
#pragma unroll
            for (int r = 0; r < 8; r++) oacc[j].f[r] *= al[r];
#pragma unroll
        for (int ks = 0; ks < 4; ks++) {
            FragAB pf;
            const u16* pp = &sP[(hloc * 16 + l16) * 136 + ks * 32 + half * 8];
            pf.q[0] = *(const uint4*)pp;
            pf.q[1] = *(const uint4*)(pp + 16);
#pragma unroll
            for (int j = 0; j < 4; j++) {
                int dim0 = wsel * 64 + j * 16;
                const u16* bp = &sVT[(dim0 + l16) * 136 + ks * 32 + half * 8];
                FragAB vf;
                vf.q[0] = *(const uint4*)bp;
                vf.q[1] = *(const uint4*)(bp + 16);
                oacc[j].v = __builtin_amdgcn_wmma_f32_16x16x32_bf16(
                    false, pf.v, false, vf.v, (short)0, oacc[j].v, false, false);
            }
        }
        __syncthreads();
    }

    // ---- normalize and store bf16 attention output ----
    float inv[8];
#pragma unroll
    for (int r = 0; r < 8; r++) inv[r] = 1.0f / sSum[hloc * 16 + half * 8 + r];
#pragma unroll
    for (int j = 0; j < 4; j++) {
        int col = h * 128 + wsel * 64 + j * 16 + l16;
#pragma unroll
        for (int r = 0; r < 8; r++) {
            int row = b * 16 + half * 8 + r;
            ob[(size_t)row * 4096 + col] = f2bf(oacc[j].f[r] * inv[r]);
        }
    }
}

// ---------------------------------------------------------------------------
// Inputs (setup_inputs order): x, cache_k, cache_v, freqs_cis, mask,
//                              wq, wk, wv, wo, start_pos
// Workspace layout (22.0 MiB total):
//   [0)         qkv  fp32 512x6144
//   [12582912)  qb   bf16 512x4096
//   [16777216)  kb   bf16 512x1024
//   [17825792)  vb   bf16 512x1024
//   [18874368)  ob   bf16 512x4096
// ---------------------------------------------------------------------------
extern "C" void kernel_launch(void* const* d_in, const int* in_sizes, int n_in,
                              void* d_out, int out_size, void* d_ws, size_t ws_size,
                              hipStream_t stream)
{
    (void)in_sizes; (void)n_in; (void)out_size; (void)ws_size;
    const float* x  = (const float*)d_in[0];
    const float* ck = (const float*)d_in[1];
    const float* cv = (const float*)d_in[2];
    const float* fr = (const float*)d_in[3];
    const u16*   wq = (const u16*)d_in[5];
    const u16*   wk = (const u16*)d_in[6];
    const u16*   wv = (const u16*)d_in[7];
    const u16*   wo = (const u16*)d_in[8];

    char* ws = (char*)d_ws;
    float* qkv = (float*)ws;
    u16*   qb  = (u16*)(ws + 12582912);
    u16*   kb  = (u16*)(ws + 16777216);
    u16*   vb  = (u16*)(ws + 17825792);
    u16*   ob  = (u16*)(ws + 18874368);

    dim3 blk(256);
    // QKV projections: x (512x4096 fp32) @ {wq,wk,wv} -> qkv fp32 512x6144
    gemm_bf16<true ><<<dim3(4, 32), blk, 0, stream>>>(x, wq, qkv,        4096, 4096, 6144, 4096);
    gemm_bf16<true ><<<dim3(4,  8), blk, 0, stream>>>(x, wk, qkv + 4096, 4096, 1024, 6144, 4096);
    gemm_bf16<true ><<<dim3(4,  8), blk, 0, stream>>>(x, wv, qkv + 5120, 4096, 1024, 6144, 4096);
    // Rotary + bf16
    rotary_convert<<<dim3(6144), blk, 0, stream>>>(qkv, fr, qb, kb, vb);
    // Flash attention: TDM-pipelined KV streaming, GQA-shared chunks
    attn_fwd<<<dim3(8, 32), blk, ATTN_LDS_BYTES, stream>>>(ck, cv, qb, kb, vb, ob);
    // Output projection -> fp32 d_out (512x4096)
    gemm_bf16<false><<<dim3(4, 32), blk, 0, stream>>>(ob, wo, (float*)d_out, 4096, 4096, 4096, 4096);
}
